// LocalIsing_60498909331881
// MI455X (gfx1250) — matrix operand order repaired and checked by reading, not compiled, not run on hardware
//
#include <hip/hip_runtime.h>
#include <hip/hip_bf16.h>

typedef _Float16 v16h __attribute__((ext_vector_type(16)));
typedef _Float16 v8h  __attribute__((ext_vector_type(8)));
typedef float    v8f  __attribute__((ext_vector_type(8)));

#define NSPIN 512
#define BATCH 1024

// ---------------------------------------------------------------------------
// Kernel 1: scatter J2 into dense symmetric f16 matrices S_hi / S_lo.
//   S_hi[i,j] = f16(J2[c]) ; S_lo[i,j] = f16(J2[c] - S_hi[i,j]) ; diag = 0
// Every off-diagonal element is written exactly once per side -> no races.
// ---------------------------------------------------------------------------
__global__ void ising_build_S(const float* __restrict__ J2,
                              const int*   __restrict__ pairs,
                              _Float16* __restrict__ S_hi,
                              _Float16* __restrict__ S_lo,
                              int C) {
    int c = blockIdx.x * blockDim.x + threadIdx.x;
    if (c < NSPIN) {                       // zero the diagonal
        S_hi[c * NSPIN + c] = (_Float16)0.f;
        S_lo[c * NSPIN + c] = (_Float16)0.f;
    }
    if (c < C) {
        int i = pairs[2 * c + 0];
        int j = pairs[2 * c + 1];
        float v  = J2[c];
        _Float16 hi = (_Float16)v;
        _Float16 lo = (_Float16)(v - (float)hi);
        S_hi[i * NSPIN + j] = hi;  S_hi[j * NSPIN + i] = hi;
        S_lo[i * NSPIN + j] = lo;  S_lo[j * NSPIN + i] = lo;
    }
}

// ---------------------------------------------------------------------------
// Kernel 2: convert spin matrix X (entries are exactly +/-1) to f16 (exact).
// ---------------------------------------------------------------------------
__global__ void ising_conv_x(const float* __restrict__ x,
                             _Float16* __restrict__ Xh, int n) {
    int i = blockIdx.x * blockDim.x + threadIdx.x;
    if (i < n) Xh[i] = (_Float16)x[i];
}

// ---------------------------------------------------------------------------
// Kernel 3: fused WMMA energy kernel.
// One wave32 per 16-row batch tile. For each 16-col n-tile, accumulate
//   C = sum_k A(16x32) * B(32x16)  via v_wmma_f32_16x16x32_f16 (hi+lo),
// then acc[v] += C[v] * x[row, n]  (elementwise Hadamard with X),
// finally half-wave reduce, add linear term x.J1, scale quadratic by 0.5.
//
// Fragment layouts used here (wave32; A/B use contiguous k, not the ISA table order):
//   A 16x32 f16 : lane l -> row m=l%16, halves K = 16*sel .. 16*sel+15
//   B 32x16 f16 : lane l -> col n=l%16, halves K = 16*sel .. 16*sel+15
//                 (S symmetric => column n == row n => contiguous loads)
//   C/D  f32    : VGPR v, lane l -> row v+8*(l>=16), col l%16
// ---------------------------------------------------------------------------
__global__ __launch_bounds__(128)
void ising_wmma_energy(const float*    __restrict__ x,
                       const float*    __restrict__ J1,
                       const _Float16* __restrict__ Xh,
                       const _Float16* __restrict__ S_hi,
                       const _Float16* __restrict__ S_lo,
                       float* __restrict__ out) {
    const int lane = threadIdx.x & 31;
    const int wave = threadIdx.x >> 5;
    const int tile = blockIdx.x * 4 + wave;     // 64 tiles of 16 batch rows
    const int rowbase = tile * 16;
    const int sel = lane >> 4;                  // 0: lanes 0-15, 1: lanes 16-31
    const int ln  = lane & 15;

    __shared__ float red[4][8][32];

    float acc[8];
#pragma unroll
    for (int v = 0; v < 8; ++v) acc[v] = 0.f;

    const _Float16* arow = Xh + (rowbase + ln) * NSPIN;   // A: row m = ln

    for (int nt = 0; nt < 32; ++nt) {
        const int nbase = nt * 16;
        const _Float16* bhrow = S_hi + (nbase + ln) * NSPIN;  // col n == row n
        const _Float16* blrow = S_lo + (nbase + ln) * NSPIN;

        v8f c = {};                                    // f32 accumulator tile
#pragma unroll 4
        for (int kc = 0; kc < NSPIN; kc += 32) {
            v8h a0 = *(const v8h*)(arow + kc + 16 * sel);
            v8h a1 = *(const v8h*)(arow + kc + 16 * sel + 8);
            v8h h0 = *(const v8h*)(bhrow + kc + 16 * sel);
            v8h h1 = *(const v8h*)(bhrow + kc + 16 * sel + 8);
            v8h l0 = *(const v8h*)(blrow + kc + 16 * sel);
            v8h l1 = *(const v8h*)(blrow + kc + 16 * sel + 8);
            v16h a, bh, bl;
#pragma unroll
            for (int t = 0; t < 8; ++t) {
                a[t] = a0[t];  a[t + 8] = a1[t];
                bh[t] = h0[t]; bh[t + 8] = h1[t];
                bl[t] = l0[t]; bl[t + 8] = l1[t];
            }
            c = __builtin_amdgcn_wmma_f32_16x16x32_f16(
                    false, a, false, bh, (short)0, c, false, false);
            c = __builtin_amdgcn_wmma_f32_16x16x32_f16(
                    false, a, false, bl, (short)0, c, false, false);
        }

        // Hadamard with X and accumulate over n: C[v]@lane = (row v+8*sel, col ln)
        const float* xcol = x + nbase + ln;
#pragma unroll
        for (int v = 0; v < 8; ++v)
            acc[v] += c[v] * xcol[(rowbase + v + 8 * sel) * NSPIN];
    }

    // Half-wave reduction via LDS: row r<8 lives in lanes 0-15 of acc[r],
    // row r>=8 in lanes 16-31 of acc[r-8].
#pragma unroll
    for (int v = 0; v < 8; ++v) red[wave][v][lane] = acc[v];
    __syncthreads();

    if (lane < 16) {
        const int r = lane;              // row within tile
        const int v = r & 7;
        const int g = r >> 3;            // lane-group holding this row
        float quad = 0.f;
#pragma unroll
        for (int l = 0; l < 16; ++l) quad += red[wave][v][g * 16 + l];

        // linear term x[row] . J1 in f32 (float4 vectorized)
        const float4* xr4 = (const float4*)(x + (rowbase + r) * NSPIN);
        const float4* j4  = (const float4*)J1;
        float lin = 0.f;
        for (int k = 0; k < NSPIN / 4; ++k) {
            float4 a = xr4[k], b = j4[k];
            lin += a.x * b.x + a.y * b.y + a.z * b.z + a.w * b.w;
        }
        out[rowbase + r] = 0.5f * quad + lin;
    }
}

// ---------------------------------------------------------------------------
// d_in order (setup_inputs): x[f32, B*N], J1[f32, N], J2[f32, C], pairs[i32, 2C]
// d_out: energy [B] f32.  Workspace layout: S_hi(512KB) | S_lo(512KB) | Xh(1MB)
// ---------------------------------------------------------------------------
extern "C" void kernel_launch(void* const* d_in, const int* in_sizes, int n_in,
                              void* d_out, int out_size, void* d_ws, size_t ws_size,
                              hipStream_t stream) {
    const float* x     = (const float*)d_in[0];
    const float* J1    = (const float*)d_in[1];
    const float* J2    = (const float*)d_in[2];
    const int*   pairs = (const int*)d_in[3];
    float* out = (float*)d_out;

    const int C = in_sizes[2];               // 130816
    const int B = in_sizes[0] / NSPIN;       // 1024

    char* ws = (char*)d_ws;
    _Float16* S_hi = (_Float16*)(ws);
    _Float16* S_lo = (_Float16*)(ws + (size_t)NSPIN * NSPIN * sizeof(_Float16));
    _Float16* Xh   = (_Float16*)(ws + (size_t)2 * NSPIN * NSPIN * sizeof(_Float16));

    {   // build symmetric split-f16 coupling matrix
        int threads = 256, blocks = (C + threads - 1) / threads;
        ising_build_S<<<blocks, threads, 0, stream>>>(J2, pairs, S_hi, S_lo, C);
    }
    {   // exact f16 conversion of spins
        int n = B * NSPIN;
        int threads = 256, blocks = (n + threads - 1) / threads;
        ising_conv_x<<<blocks, threads, 0, stream>>>(x, Xh, n);
    }
    {   // fused WMMA energy: 64 tiles, 4 waves/block
        int tiles = B / 16;
        ising_wmma_energy<<<tiles / 4, 128, 0, stream>>>(x, J1, Xh, S_hi, S_lo, out);
    }
}
